// HeadExtractor_89953795047565
// MI455X (gfx1250) — matrix operand order repaired and checked by reading, compiled, not run on hardware
//
#include <hip/hip_runtime.h>
#include <hip/hip_bf16.h>
#include <math.h>

// ---------------------------------------------------------------------------
// GATv2 (2 layers, D=128, H=4, DH=32) + pooling + 2 LN-MLP heads for gfx1250.
// GEMMs run on v_wmma_f32_16x16x32_f16 (f16 in, f32 acc). Edge phase is
// wave32-per-edge with L2-resident gathers and atomic scatters.
// ---------------------------------------------------------------------------

typedef __attribute__((ext_vector_type(16))) _Float16 v16h;
typedef __attribute__((ext_vector_type(8)))  float    v8f;

#define DIM 128
#define NHEAD 4
#define DHEAD 32

// ---- monotone uint key for float atomic max ----
__device__ __forceinline__ unsigned f2key(float f) {
    unsigned ib = __float_as_uint(f);
    return (ib & 0x80000000u) ? ~ib : (ib | 0x80000000u);
}
__device__ __forceinline__ float key2f(unsigned key) {
    unsigned ib = (key & 0x80000000u) ? (key ^ 0x80000000u) : ~key;
    return __uint_as_float(ib);
}

// ---------------------------------------------------------------------------
// node_to_local scatter (ntl pre-memset to -1)
__global__ void build_ntl_kernel(const int* __restrict__ subset, int* __restrict__ ntl, int S) {
    int i = blockIdx.x * blockDim.x + threadIdx.x;
    if (i < S) ntl[subset[i]] = i;
}

// Build remapped edge list + validity; tail entries are self loops.
__global__ void build_edges_kernel(const int* __restrict__ ei, const int* __restrict__ ntl,
                                   int* __restrict__ esrc, int* __restrict__ edst,
                                   int* __restrict__ evalid, int E, int S) {
    int e = blockIdx.x * blockDim.x + threadIdx.x;
    int Etot = E + S;
    if (e >= Etot) return;
    if (e < E) {
        int sl = ntl[ei[e]];
        int dl = ntl[ei[E + e]];
        int v = (sl >= 0) & (dl >= 0) & (sl != dl);
        esrc[e] = sl < 0 ? 0 : sl;
        edst[e] = dl < 0 ? 0 : dl;
        evalid[e] = v;
    } else {
        int i = e - E;
        esrc[e] = i; edst[e] = i; evalid[e] = 1;
    }
}

// Gather x = node_embeddings[subset] ; elementwise
__global__ void gather_x_kernel(const float* __restrict__ emb, const int* __restrict__ subset,
                                float* __restrict__ x, int S) {
    long long i = (long long)blockIdx.x * blockDim.x + threadIdx.x;
    if (i >= (long long)S * DIM) return;
    int row = (int)(i >> 7), d = (int)(i & 127);
    x[i] = emb[(long long)subset[row] * DIM + d];
}

// f32 -> f16 activation convert
__global__ void to_half_kernel(const float* __restrict__ x, _Float16* __restrict__ xh, long long n) {
    long long i = (long long)blockIdx.x * blockDim.x + threadIdx.x;
    if (i < n) xh[i] = (_Float16)x[i];
}

// Weight transpose + f16 convert: Wt[n*128+k] = W[k*128+n]
__global__ void weight_t_h_kernel(const float* __restrict__ W, _Float16* __restrict__ Wt) {
    int i = blockIdx.x * blockDim.x + threadIdx.x;
    if (i >= DIM * DIM) return;
    int n = i >> 7, k = i & 127;
    Wt[i] = (_Float16)W[k * DIM + n];
}

// ---------------------------------------------------------------------------
// WMMA GEMM: Out[S,128] = Xh[S,128] (f16, row-major) @ W (via Wt[n][k] f16).
// One wave computes one 16x16 tile, K swept in 4 steps of 32.
// Operand layouts per CDNA5 ISA 7.12.2 (16-bit A 16x32; B mirrored; f32 C/D).
__global__ void gemm_wmma_kernel(const _Float16* __restrict__ Xh,
                                 const _Float16* __restrict__ Wt,
                                 float* __restrict__ Out, int Srows) {
    int wid  = blockIdx.x * (blockDim.x >> 5) + (threadIdx.x >> 5);
    int lane = threadIdx.x & 31;
    int tileM = wid >> 3;          // 8 N-tiles (128/16)
    int tileN = wid & 7;
    if (tileM * 16 >= Srows) return;

    int mrow = tileM * 16 + (lane & 15);
    int ncol = tileN * 16 + (lane & 15);
    int hi   = lane >> 4;          // lane-half selector

    v8f c = {};
    for (int kb = 0; kb < DIM; kb += 32) {
        v16h a, b;
        // A fragment: lanes<16 -> K kb+[0..7] / kb+16+[0..7]; lanes>=16 shifted by 8
        const _Float16* ap = Xh + (long long)mrow * DIM + kb + hi * 8;
#pragma unroll
        for (int i = 0; i < 8; ++i) { a[i] = ap[i]; a[8 + i] = ap[16 + i]; }
        // B fragment: lanes<16 -> K kb+[0..7] / kb+16+[0..7]; lanes>=16 shifted by 8, col = lane&15
        const _Float16* bp = Wt + (long long)ncol * DIM + kb + hi * 8;
#pragma unroll
        for (int i = 0; i < 8; ++i) { b[i] = bp[i]; b[8 + i] = bp[16 + i]; }
        c = __builtin_amdgcn_wmma_f32_16x16x32_f16(false, a, false, b, (short)0, c, false, false);
    }
    // C/D: VGPR v -> M = v + 8*hi ; N = lane&15
#pragma unroll
    for (int v = 0; v < 8; ++v) {
        int m = tileM * 16 + v + 8 * hi;
        Out[(long long)m * DIM + ncol] = c[v];
    }
}

// ---------------------------------------------------------------------------
// Pass 1: per-edge attention logits + segment max (keyed uint atomics).
// One wave32 per edge; lane L covers d = h*32+L for h = 0..3.
__global__ void edge_logits_kernel(const float* __restrict__ xl, const float* __restrict__ xr,
                                   const int* __restrict__ esrc, const int* __restrict__ edst,
                                   const int* __restrict__ evalid, const float* __restrict__ att,
                                   float* __restrict__ logits, unsigned* __restrict__ mx, int Etot) {
    int wid = blockIdx.x * (blockDim.x >> 5) + (threadIdx.x >> 5);
    if (wid >= Etot) return;
    if (!evalid[wid]) return;
    int lane = threadIdx.x & 31;
    int s = esrc[wid], d = edst[wid];
    float acc[NHEAD];
#pragma unroll
    for (int h = 0; h < NHEAD; ++h) {
        int idx = h * DHEAD + lane;
        float v = xl[(long long)s * DIM + idx] + xr[(long long)d * DIM + idx];
        v = v > 0.f ? v : 0.2f * v;           // leaky_relu(0.2)
        acc[h] = v * att[idx];
    }
#pragma unroll
    for (int off = 16; off > 0; off >>= 1) {
#pragma unroll
        for (int h = 0; h < NHEAD; ++h) acc[h] += __shfl_xor(acc[h], off, 32);
    }
    if (lane < NHEAD) {
        float lg = acc[lane];
        logits[(long long)wid * NHEAD + lane] = lg;
        atomicMax(&mx[(long long)d * NHEAD + lane], f2key(lg));
    }
}

// Pass 2: ex = exp(logit - mx[dst]); accumulate denominators. Thread per edge.
__global__ void edge_exp_kernel(const int* __restrict__ edst, const int* __restrict__ evalid,
                                float* __restrict__ logits, const unsigned* __restrict__ mx,
                                float* __restrict__ den, int Etot) {
    int e = blockIdx.x * blockDim.x + threadIdx.x;
    if (e >= Etot) return;
    if (!evalid[e]) return;
    int d = edst[e];
#pragma unroll
    for (int h = 0; h < NHEAD; ++h) {
        float m  = key2f(mx[(long long)d * NHEAD + h]);
        float ex = __expf(logits[(long long)e * NHEAD + h] - m);
        logits[(long long)e * NHEAD + h] = ex;
        atomicAdd(&den[(long long)d * NHEAD + h], ex);
    }
}

// agg[i][d] = bias[d] (segment_sum result seeded with bias)
__global__ void init_agg_kernel(float* __restrict__ agg, const float* __restrict__ bias, int S) {
    long long i = (long long)blockIdx.x * blockDim.x + threadIdx.x;
    if (i >= (long long)S * DIM) return;
    agg[i] = bias[(int)(i & 127)];
}

// Pass 3: weighted scatter of xl[src] into agg[dst]. One wave32 per edge.
__global__ void edge_scatter_kernel(const float* __restrict__ xl, const float* __restrict__ logits,
                                    const float* __restrict__ den, const int* __restrict__ esrc,
                                    const int* __restrict__ edst, const int* __restrict__ evalid,
                                    float* __restrict__ agg, int Etot) {
    int wid = blockIdx.x * (blockDim.x >> 5) + (threadIdx.x >> 5);
    if (wid >= Etot) return;
    if (!evalid[wid]) return;
    int lane = threadIdx.x & 31;
    int s = esrc[wid], d = edst[wid];
#pragma unroll
    for (int h = 0; h < NHEAD; ++h) {
        float w = logits[(long long)wid * NHEAD + h] /
                  (den[(long long)d * NHEAD + h] + 1e-16f);
        int idx = h * DHEAD + lane;
        atomicAdd(&agg[(long long)d * DIM + idx], w * xl[(long long)s * DIM + idx]);
    }
}

// x = relu(layer_norm(agg + x, g, b)); one wave32 per row (4 elems/lane).
__global__ void post_ln_kernel(const float* __restrict__ agg, float* __restrict__ x,
                               const float* __restrict__ g, const float* __restrict__ b, int Srows) {
    int row = blockIdx.x * (blockDim.x >> 5) + (threadIdx.x >> 5);
    if (row >= Srows) return;
    int lane = threadIdx.x & 31;
    float v[4];
    float sum = 0.f;
#pragma unroll
    for (int i = 0; i < 4; ++i) {
        long long idx = (long long)row * DIM + i * 32 + lane;
        v[i] = agg[idx] + x[idx];
        sum += v[i];
    }
#pragma unroll
    for (int off = 16; off > 0; off >>= 1) sum += __shfl_xor(sum, off, 32);
    float mean = sum * (1.f / DIM);
    float var = 0.f;
#pragma unroll
    for (int i = 0; i < 4; ++i) { float dv = v[i] - mean; var += dv * dv; }
#pragma unroll
    for (int off = 16; off > 0; off >>= 1) var += __shfl_xor(var, off, 32);
    float inv = rsqrtf(var * (1.f / DIM) + 1e-5f);
#pragma unroll
    for (int i = 0; i < 4; ++i) {
        int d = i * 32 + lane;
        float y = (v[i] - mean) * inv * g[d] + b[d];
        x[(long long)row * DIM + d] = fmaxf(y, 0.f);
    }
}

// ---------------------------------------------------------------------------
// Pooling: per-block partial sum/max over a row chunk (128 threads = columns).
__global__ void pool_partial_kernel(const float* __restrict__ x, float* __restrict__ psum,
                                    unsigned* __restrict__ pmaxkey, int Srows) {
    int c  = threadIdx.x;                 // 128 threads
    int r0 = blockIdx.x * 256;
    int r1 = r0 + 256; if (r1 > Srows) r1 = Srows;
    float s = 0.f, m = -3.4e38f;
    for (int r = r0; r < r1; ++r) {
        float v = x[(long long)r * DIM + c];
        s += v; m = fmaxf(m, v);
    }
    atomicAdd(&psum[c], s);
    atomicMax(&pmaxkey[c], f2key(m));
}

__global__ void pool_finalize_kernel(const float* __restrict__ psum,
                                     const unsigned* __restrict__ pmaxkey,
                                     float* __restrict__ combined, int Srows) {
    int c = threadIdx.x;
    if (c >= DIM) return;
    float s = psum[c];
    combined[c]           = s / (float)Srows;   // mean
    combined[DIM + c]     = key2f(pmaxkey[c]);  // max
    combined[2 * DIM + c] = s;                  // sum
}

// out[j] = bias[j] + sum_i vec[i] * W[i*Out + j]
__global__ void matvec_kernel(const float* __restrict__ vec, const float* __restrict__ W,
                              const float* __restrict__ bias, float* __restrict__ out,
                              int In, int Outn) {
    int j = blockIdx.x * blockDim.x + threadIdx.x;
    if (j >= Outn) return;
    float acc = bias[j];
    for (int i = 0; i < In; ++i) acc += vec[i] * W[(long long)i * Outn + j];
    out[j] = acc;
}

// Single-block LN + ReLU over an n-vector (n = blockDim.x, power of two).
__global__ void ln_relu_vec_kernel(const float* __restrict__ t, const float* __restrict__ g,
                                   const float* __restrict__ b, float* __restrict__ out, int n) {
    __shared__ float red[512];
    int tid = threadIdx.x;
    float v = t[tid];
    red[tid] = v; __syncthreads();
    for (int s = blockDim.x >> 1; s > 0; s >>= 1) {
        if (tid < s) red[tid] += red[tid + s];
        __syncthreads();
    }
    float mean = red[0] / (float)n; __syncthreads();
    float dv = v - mean;
    red[tid] = dv * dv; __syncthreads();
    for (int s = blockDim.x >> 1; s > 0; s >>= 1) {
        if (tid < s) red[tid] += red[tid + s];
        __syncthreads();
    }
    float inv = rsqrtf(red[0] / (float)n + 1e-5f);
    out[tid] = fmaxf(dv * inv * g[tid] + b[tid], 0.f);
}

// ---------------------------------------------------------------------------
extern "C" void kernel_launch(void* const* d_in, const int* in_sizes, int n_in,
                              void* d_out, int out_size, void* d_ws, size_t ws_size,
                              hipStream_t stream) {
    const float* emb    = (const float*)d_in[0];
    const int*   subset = (const int*)  d_in[1];
    const int*   ei     = (const int*)  d_in[2];
    // d_in[3] = batch, unused
    const float* lin_l[2] = { (const float*)d_in[4],  (const float*)d_in[10] };
    const float* lin_r[2] = { (const float*)d_in[5],  (const float*)d_in[11] };
    const float* attp[2]  = { (const float*)d_in[6],  (const float*)d_in[12] };
    const float* biasp[2] = { (const float*)d_in[7],  (const float*)d_in[13] };
    const float* gp[2]    = { (const float*)d_in[8],  (const float*)d_in[14] };
    const float* bp[2]    = { (const float*)d_in[9],  (const float*)d_in[15] };
    const float* W1  = (const float*)d_in[16]; const float* bW1 = (const float*)d_in[17];
    const float* ga1 = (const float*)d_in[18]; const float* ba1 = (const float*)d_in[19];
    const float* W2  = (const float*)d_in[20]; const float* bW2 = (const float*)d_in[21];
    const float* ga2 = (const float*)d_in[22]; const float* ba2 = (const float*)d_in[23];

    const int Nn   = in_sizes[0] / DIM;
    const int S    = in_sizes[1];
    const int E    = in_sizes[2] / 2;
    const int Etot = E + S;

    // ---- workspace carve-up (256B aligned) ----
    char* base = (char*)d_ws;
    size_t off = 0;
    auto carve = [&](size_t bytes) { void* p = base + off; off = (off + bytes + 255) & ~(size_t)255; return p; };
    int*       ntl     = (int*)      carve((size_t)Nn * 4);
    int*       esrc    = (int*)      carve((size_t)Etot * 4);
    int*       edst    = (int*)      carve((size_t)Etot * 4);
    int*       evalid  = (int*)      carve((size_t)Etot * 4);
    float*     x       = (float*)    carve((size_t)S * DIM * 4);
    _Float16*  Xh      = (_Float16*) carve((size_t)S * DIM * 2);
    float*     xl      = (float*)    carve((size_t)S * DIM * 4);
    float*     xragg   = (float*)    carve((size_t)S * DIM * 4);   // xr, then reused as agg
    float*     logits  = (float*)    carve((size_t)Etot * NHEAD * 4);
    unsigned*  mx      = (unsigned*) carve((size_t)S * NHEAD * 4);
    float*     den     = (float*)    carve((size_t)S * NHEAD * 4);
    _Float16*  Wt[4];
    for (int i = 0; i < 4; ++i) Wt[i] = (_Float16*)carve((size_t)DIM * DIM * 2);
    float*     psum    = (float*)    carve(DIM * 4);
    unsigned*  pmaxkey = (unsigned*) carve(DIM * 4);
    float*     combined= (float*)    carve(3 * DIM * 4);
    float*     t1      = (float*)    carve(512 * 4);
    float*     h1      = (float*)    carve(512 * 4);
    float*     t2      = (float*)    carve(256 * 4);
    (void)ws_size; (void)n_in; (void)out_size;

    const int TB = 256;
    const long long SD = (long long)S * DIM;

    // ---- graph preprocessing ----
    hipMemsetAsync(ntl, 0xFF, (size_t)Nn * 4, stream);             // -1
    build_ntl_kernel<<<(S + TB - 1) / TB, TB, 0, stream>>>(subset, ntl, S);
    build_edges_kernel<<<(Etot + TB - 1) / TB, TB, 0, stream>>>(ei, ntl, esrc, edst, evalid, E, S);

    // ---- weights -> f16 transposed ----
    weight_t_h_kernel<<<(DIM * DIM + TB - 1) / TB, TB, 0, stream>>>(lin_l[0], Wt[0]);
    weight_t_h_kernel<<<(DIM * DIM + TB - 1) / TB, TB, 0, stream>>>(lin_r[0], Wt[1]);
    weight_t_h_kernel<<<(DIM * DIM + TB - 1) / TB, TB, 0, stream>>>(lin_l[1], Wt[2]);
    weight_t_h_kernel<<<(DIM * DIM + TB - 1) / TB, TB, 0, stream>>>(lin_r[1], Wt[3]);

    // ---- gather subset rows ----
    gather_x_kernel<<<(int)((SD + TB - 1) / TB), TB, 0, stream>>>(emb, subset, x, S);

    const int gemmBlocks = S / 16;                 // 8 waves/block = 8 N-tiles of one M-tile
    const int edgeWaveBlocks = (Etot + 7) / 8;     // 8 waves of 32 per 256-thread block
    const int rowWaveBlocks  = (S + 7) / 8;

    for (int L = 0; L < 2; ++L) {
        to_half_kernel<<<(int)((SD + TB - 1) / TB), TB, 0, stream>>>(x, Xh, SD);
        gemm_wmma_kernel<<<gemmBlocks, TB, 0, stream>>>(Xh, Wt[2 * L + 0], xl,    S);
        gemm_wmma_kernel<<<gemmBlocks, TB, 0, stream>>>(Xh, Wt[2 * L + 1], xragg, S);

        hipMemsetAsync(mx,  0, (size_t)S * NHEAD * 4, stream);     // key 0 == -inf
        hipMemsetAsync(den, 0, (size_t)S * NHEAD * 4, stream);

        edge_logits_kernel<<<edgeWaveBlocks, TB, 0, stream>>>(xl, xragg, esrc, edst, evalid,
                                                              attp[L], logits, mx, Etot);
        edge_exp_kernel<<<(Etot + TB - 1) / TB, TB, 0, stream>>>(edst, evalid, logits, mx, den, Etot);
        init_agg_kernel<<<(int)((SD + TB - 1) / TB), TB, 0, stream>>>(xragg, biasp[L], S);
        edge_scatter_kernel<<<edgeWaveBlocks, TB, 0, stream>>>(xl, logits, den, esrc, edst,
                                                               evalid, xragg, Etot);
        post_ln_kernel<<<rowWaveBlocks, TB, 0, stream>>>(xragg, x, gp[L], bp[L], S);
    }

    // ---- pooling ----
    hipMemsetAsync(psum, 0, DIM * 4, stream);
    hipMemsetAsync(pmaxkey, 0, DIM * 4, stream);
    pool_partial_kernel<<<(S + 255) / 256, DIM, 0, stream>>>(x, psum, pmaxkey, S);
    pool_finalize_kernel<<<1, DIM, 0, stream>>>(psum, pmaxkey, combined, S);

    // ---- MLP heads ----
    matvec_kernel<<<2, 256, 0, stream>>>(combined, W1, bW1, t1, 3 * DIM, 512);
    ln_relu_vec_kernel<<<1, 512, 0, stream>>>(t1, ga1, ba1, h1, 512);
    matvec_kernel<<<1, 256, 0, stream>>>(h1, W2, bW2, t2, 512, 256);
    ln_relu_vec_kernel<<<1, 256, 0, stream>>>(t2, ga2, ba2, (float*)d_out, 256);
}
